// NeuralCDE_24644522344752
// MI455X (gfx1250) — matrix operand order repaired and checked by reading, not compiled, not run on hardware
//
#include <hip/hip_runtime.h>
#include <hip/hip_bf16.h>
#include <math.h>

typedef __attribute__((ext_vector_type(16))) _Float16 v16h;
typedef __attribute__((ext_vector_type(8)))  _Float16 v8h;
typedef __attribute__((ext_vector_type(8)))  float    v8f;

#define T_SAVE 2048
#define CTRL   8
#define HID    128
#define WID    512
#define NSUB   4
#define K0P    160   // 136 padded to multiple of 32
#define KI0P   32    // 8 padded to 32

// Tsit5 tableau
__device__ __constant__ float d_A21 = 0.161f;
__device__ __constant__ float d_A31 = -0.008480655492356989f, d_A32 = 0.335480655492357f;
__device__ __constant__ float d_A41 = 2.8971530571054935f,  d_A42 = -6.359448489975075f, d_A43 = 4.3622954328695815f;
__device__ __constant__ float d_A51 = 5.325864828439257f,   d_A52 = -11.748883564062828f, d_A53 = 7.4955393428898365f, d_A54 = -0.09249506636175525f;
__device__ __constant__ float d_A61 = 5.86145544294642f,    d_A62 = -12.92096931784711f,  d_A63 = 8.159367898576159f,  d_A64 = -0.071584973281401f, d_A65 = -0.028269050394068383f;
__device__ __constant__ float d_B1 = 0.09646076681806523f, d_B2 = 0.01f, d_B3 = 0.4798896504144996f,
                              d_B4 = 1.379008574103742f,   d_B5 = -3.290069515436081f, d_B6 = 2.324710524099774f;
__device__ __constant__ float d_C2 = 0.161f, d_C3 = 0.327f, d_C4 = 0.9f, d_C5 = 0.9800255409045097f;

// ---------------- weight conversion (f32 row-major -> f16 row-major, K padded) ----------------
__global__ void cvt_pad_kernel(const float* __restrict__ src, _Float16* __restrict__ dst,
                               int rows, int K, int Kp) {
    int i = blockIdx.x * blockDim.x + threadIdx.x;
    int total = rows * Kp;
    if (i >= total) return;
    int r = i / Kp, c = i - r * Kp;
    dst[i] = (c < K) ? (_Float16)src[r * K + c] : (_Float16)0.f;
}

// ---------------- device helpers ----------------
__device__ __forceinline__ float act_apply(float x, int act) {
    if (act == 0) return (x > 20.f) ? x : log1pf(__expf(x)); // softplus
    if (act == 1) return fmaxf(x, 0.f);                      // relu
    if (act == 2) return tanhf(x);                           // tanh
    return x;                                                // identity
}

// Load one 16x32 f16 A fragment (row-major weights) + broadcast B chunk, WMMA-accumulate.
__device__ __forceinline__ v8f wmma_acc_chunk(const _Float16* rowp, const _Float16* hin,
                                              int kc, int hi, v8f acc) {
    union { v16h v; v8h h[2]; } A, B;
    A.h[0] = *(const v8h*)(rowp + kc * 32);
    A.h[1] = *(const v8h*)(rowp + kc * 32 + 16);
    const _Float16* bp = hin + kc * 32 + hi * 8;
    B.h[0] = *(const v8h*)(bp);
    B.h[1] = *(const v8h*)(bp + 16);
    return __builtin_amdgcn_wmma_f32_16x16x32_f16(false, A.v, false, B.v,
                                                  (short)0, acc, false, false);
}

// Hint-prefetch one wave's A-tile rows for an upcoming layer (global_prefetch_b8).
__device__ __forceinline__ void prefetch_rows(const _Float16* __restrict__ W, int Kp, int tile) {
    const int lane = threadIdx.x & 31;
    const int lm = lane & 15, hi = lane >> 4;
    const _Float16* rowp = W + (size_t)(tile * 16 + lm) * (size_t)Kp + hi * 8;
    const int nk = Kp >> 5;
    for (int kc = 0; kc < nk; kc += 2)   // one hint per ~128B along the row
        __builtin_prefetch((const void*)(rowp + kc * 32), 0, 0);
}

// y = act(W @ hin + bias); one 16-row tile per wave; dual accumulators for matrix-pipe ILP.
__device__ __forceinline__ void wmma_layer(const _Float16* __restrict__ W,
                                           const float* __restrict__ bias,
                                           const _Float16* hin,
                                           _Float16* hout, float* fout,
                                           int rows, int Kp, int act) {
    const int tid  = threadIdx.x;
    const int wave = tid >> 5;
    const int lane = tid & 31;
    const int lm   = lane & 15;   // C/D: N index
    const int hi   = lane >> 4;
    const int ntiles = rows >> 4;
    for (int tile = wave; tile < ntiles; tile += 32) {
        v8f acc0 = {0.f, 0.f, 0.f, 0.f, 0.f, 0.f, 0.f, 0.f};
        v8f acc1 = {0.f, 0.f, 0.f, 0.f, 0.f, 0.f, 0.f, 0.f};
        const _Float16* rowp = W + (size_t)(tile * 16 + lm) * (size_t)Kp + hi * 8;
        const int nk = Kp >> 5;
        for (int kc = 0; kc < nk; kc += 2) {
            acc0 = wmma_acc_chunk(rowp, hin, kc, hi, acc0);
            if (kc + 1 < nk) acc1 = wmma_acc_chunk(rowp, hin, kc + 1, hi, acc1);
        }
        v8f acc = acc0 + acc1;
        // every column identical; lanes with N==0 (lane 0 -> M=0..7, lane 16 -> M=8..15) write out
        if (lm == 0) {
#pragma unroll
            for (int v = 0; v < 8; ++v) {
                int r = tile * 16 + hi * 8 + v;
                float x = acc[v] + bias[r];
                x = act_apply(x, act);
                if (hout) hout[r] = (_Float16)x;
                if (fout) fout[r] = x;
            }
        }
    }
}

// 128-row layer (Kp=512): all 32 waves participate, K split 4-way across waves,
// partials reduced through LDS. Chain length per wave: 2 WMMAs.
__device__ __forceinline__ void wmma_layer_ksplit(const _Float16* __restrict__ W,
                                                  const float* __restrict__ bias,
                                                  const _Float16* hin,
                                                  _Float16* hout, float* fout,
                                                  float* psum, int act) {
    const int tid  = threadIdx.x;
    const int wave = tid >> 5;
    const int lane = tid & 31;
    const int lm   = lane & 15;
    const int hi   = lane >> 4;
    const int tile = wave >> 2;   // 0..7
    const int ks   = wave & 3;    // covers kc = ks*4 .. ks*4+3
    v8f acc0 = {0.f, 0.f, 0.f, 0.f, 0.f, 0.f, 0.f, 0.f};
    v8f acc1 = {0.f, 0.f, 0.f, 0.f, 0.f, 0.f, 0.f, 0.f};
    const _Float16* rowp = W + (size_t)(tile * 16 + lm) * (size_t)WID + hi * 8;
    acc0 = wmma_acc_chunk(rowp, hin, ks * 4 + 0, hi, acc0);
    acc1 = wmma_acc_chunk(rowp, hin, ks * 4 + 1, hi, acc1);
    acc0 = wmma_acc_chunk(rowp, hin, ks * 4 + 2, hi, acc0);
    acc1 = wmma_acc_chunk(rowp, hin, ks * 4 + 3, hi, acc1);
    v8f acc = acc0 + acc1;
    if (lm == 0) {
#pragma unroll
        for (int v = 0; v < 8; ++v)
            psum[(tile * 16 + hi * 8 + v) * 4 + ks] = acc[v];
    }
    __syncthreads();
    if (tid < HID) {
        float x = psum[tid * 4 + 0] + psum[tid * 4 + 1] +
                  psum[tid * 4 + 2] + psum[tid * 4 + 3] + bias[tid];
        x = act_apply(x, act);
        if (hout) hout[tid] = (_Float16)x;
        if (fout) fout[tid] = x;
    }
}

__device__ __forceinline__ float ctrl_eval_c(float t, int i, int c,
                                             const float* __restrict__ ts,
                                             const float* __restrict__ cd,
                                             const float* __restrict__ cc,
                                             const float* __restrict__ cb,
                                             const float* __restrict__ ca) {
    int idx = i;
    if ((i + 1 < T_SAVE - 1) && (t >= ts[i + 1])) idx = i + 1;  // searchsorted-right, clamped
    float x  = t - ts[idx];
    float vd = cd[idx * CTRL + c], vc = cc[idx * CTRL + c];
    float vb = cb[idx * CTRL + c], va = ca[idx * CTRL + c];
    return va + x * (vb + x * (vc + x * vd));
}

// ---------------- persistent single-workgroup solver ----------------
__launch_bounds__(1024, 1)
__global__ void cde_solver(const float* __restrict__ ts,
                           const float* __restrict__ cd, const float* __restrict__ cc,
                           const float* __restrict__ cb, const float* __restrict__ ca,
                           const _Float16* __restrict__ Wi0p, const float* __restrict__ bi0,
                           const _Float16* __restrict__ Wi1p, const float* __restrict__ bi1,
                           const _Float16* __restrict__ Wi2p, const float* __restrict__ bi2,
                           const _Float16* __restrict__ Wf0p, const float* __restrict__ bf0,
                           const _Float16* __restrict__ Wf1p, const float* __restrict__ bf1,
                           const _Float16* __restrict__ Wf2p, const float* __restrict__ bf2,
                           const float* __restrict__ Wl, const float* __restrict__ bl,
                           float* __restrict__ out) {
    __shared__ __align__(16) _Float16 zin[K0P];
    __shared__ __align__(16) _Float16 hA[WID];
    __shared__ __align__(16) _Float16 hB[WID];
    __shared__ float y[HID];
    __shared__ float kb[6][HID];
    __shared__ float psum[HID * 4];

    const int tid  = threadIdx.x;
    const int wave = tid >> 5;

    // yval is meaningful for tid < HID; fills the padded MLP input and control channels
    auto finish_z = [&](float yval, float tstage, int i) {
        if (tid < HID)                zin[tid] = (_Float16)yval;
        else if (tid < HID + CTRL)    zin[tid] = (_Float16)ctrl_eval_c(tstage, i, tid - HID, ts, cd, cc, cb, ca);
        else if (tid < K0P)           zin[tid] = (_Float16)0.f;
        __syncthreads();
    };

    auto vf = [&](float* kout) {
        prefetch_rows(Wf1p, WID, wave);                         // warm next layer under L0 math
        wmma_layer(Wf0p, bf0, zin, hA, nullptr, WID, K0P, 0);
        prefetch_rows(Wf2p, WID, wave >> 2);                    // warm output layer under L1 math
        __syncthreads();
        wmma_layer(Wf1p, bf1, hA, hB, nullptr, WID, WID, 0);
        __syncthreads();
        wmma_layer_ksplit(Wf2p, bf2, hB, nullptr, kout, psum, 2);
        __syncthreads();
    };

    // ---- init MLP: y0 = relu-MLP(ctrl(ts[0])) ----
    {
        if (tid < CTRL)       zin[tid] = (_Float16)ctrl_eval_c(ts[0], 0, tid, ts, cd, cc, cb, ca);
        else if (tid < KI0P)  zin[tid] = (_Float16)0.f;
        __syncthreads();
        wmma_layer(Wi0p, bi0, zin, hA, nullptr, WID, KI0P, 1); __syncthreads();
        wmma_layer(Wi1p, bi1, hA, hB, nullptr, WID, WID, 1);   __syncthreads();
        wmma_layer_ksplit(Wi2p, bi2, hB, nullptr, y, psum, 3); __syncthreads();
        if (tid < 3) {
            float s = bl[tid];
            for (int j = 0; j < HID; ++j) s += Wl[tid * HID + j] * y[j];
            out[tid] = s;
        }
        __syncthreads();
    }

    // ---- sequential Tsit5 scan over save intervals ----
    for (int i = 0; i < T_SAVE - 1; ++i) {
        const float t0 = ts[i], t1 = ts[i + 1];
        const float h = (t1 - t0) * (1.f / NSUB);
        for (int j = 0; j < NSUB; ++j) {
            const float t = t0 + (float)j * h;

            // stage 1
            { float yv = (tid < HID) ? y[tid] : 0.f;
              finish_z(yv, t, i); vf(kb[0]); }
            // stage 2
            { float yv = 0.f;
              if (tid < HID) yv = y[tid] + h * (d_A21 * kb[0][tid]);
              finish_z(yv, t + d_C2 * h, i); vf(kb[1]); }
            // stage 3
            { float yv = 0.f;
              if (tid < HID) yv = y[tid] + h * (d_A31 * kb[0][tid] + d_A32 * kb[1][tid]);
              finish_z(yv, t + d_C3 * h, i); vf(kb[2]); }
            // stage 4
            { float yv = 0.f;
              if (tid < HID) yv = y[tid] + h * (d_A41 * kb[0][tid] + d_A42 * kb[1][tid] + d_A43 * kb[2][tid]);
              finish_z(yv, t + d_C4 * h, i); vf(kb[3]); }
            // stage 5
            { float yv = 0.f;
              if (tid < HID) yv = y[tid] + h * (d_A51 * kb[0][tid] + d_A52 * kb[1][tid] +
                                                d_A53 * kb[2][tid] + d_A54 * kb[3][tid]);
              finish_z(yv, t + d_C5 * h, i); vf(kb[4]); }
            // stage 6
            { float yv = 0.f;
              if (tid < HID) yv = y[tid] + h * (d_A61 * kb[0][tid] + d_A62 * kb[1][tid] +
                                                d_A63 * kb[2][tid] + d_A64 * kb[3][tid] +
                                                d_A65 * kb[4][tid]);
              finish_z(yv, t + h, i); vf(kb[5]); }

            // 5th-order update
            if (tid < HID)
                y[tid] += h * (d_B1 * kb[0][tid] + d_B2 * kb[1][tid] + d_B3 * kb[2][tid] +
                               d_B4 * kb[3][tid] + d_B5 * kb[4][tid] + d_B6 * kb[5][tid]);
            __syncthreads();
        }
        // save point: out[i+1] = Wl @ y + bl
        if (tid < 3) {
            float s = bl[tid];
            for (int j = 0; j < HID; ++j) s += Wl[tid * HID + j] * y[j];
            out[(i + 1) * 3 + tid] = s;
        }
        __syncthreads();
    }
}

// ---------------- host entry ----------------
extern "C" void kernel_launch(void* const* d_in, const int* in_sizes, int n_in,
                              void* d_out, int out_size, void* d_ws, size_t ws_size,
                              hipStream_t stream) {
    (void)in_sizes; (void)n_in; (void)out_size; (void)ws_size;
    const float* ts  = (const float*)d_in[0];
    const float* cd  = (const float*)d_in[1];
    const float* cc  = (const float*)d_in[2];
    const float* cb  = (const float*)d_in[3];
    const float* ca  = (const float*)d_in[4];
    const float* Wi0 = (const float*)d_in[5];
    const float* bi0 = (const float*)d_in[6];
    const float* Wi1 = (const float*)d_in[7];
    const float* bi1 = (const float*)d_in[8];
    const float* Wi2 = (const float*)d_in[9];
    const float* bi2 = (const float*)d_in[10];
    const float* Wf0 = (const float*)d_in[11];
    const float* bf0 = (const float*)d_in[12];
    const float* Wf1 = (const float*)d_in[13];
    const float* bf1 = (const float*)d_in[14];
    const float* Wf2 = (const float*)d_in[15];
    const float* bf2 = (const float*)d_in[16];
    const float* Wl  = (const float*)d_in[17];
    const float* bl  = (const float*)d_in[18];

    // f16 padded weight images in workspace (halves)
    _Float16* ws   = (_Float16*)d_ws;
    _Float16* Wi0p = ws + 0;        // 512*32   = 16384
    _Float16* Wi1p = ws + 16384;    // 512*512  = 262144
    _Float16* Wi2p = ws + 278528;   // 128*512  = 65536
    _Float16* Wf0p = ws + 344064;   // 512*160  = 81920
    _Float16* Wf1p = ws + 425984;   // 512*512  = 262144
    _Float16* Wf2p = ws + 688128;   // 128*512  = 65536  (total 753664 halves ~ 1.5 MB)

    auto cvt = [&](const float* s, _Float16* d, int rows, int K, int Kp) {
        int tot = rows * Kp;
        cvt_pad_kernel<<<(tot + 255) / 256, 256, 0, stream>>>(s, d, rows, K, Kp);
    };
    cvt(Wi0, Wi0p, 512, 8, 32);
    cvt(Wi1, Wi1p, 512, 512, 512);
    cvt(Wi2, Wi2p, 128, 512, 512);
    cvt(Wf0, Wf0p, 512, 136, 160);
    cvt(Wf1, Wf1p, 512, 512, 512);
    cvt(Wf2, Wf2p, 128, 512, 512);

    cde_solver<<<1, 1024, 0, stream>>>(ts, cd, cc, cb, ca,
                                       Wi0p, bi0, Wi1p, bi1, Wi2p, bi2,
                                       Wf0p, bf0, Wf1p, bf1, Wf2p, bf2,
                                       Wl, bl, (float*)d_out);
}
